// MultiSkillPolicyNet1_35330400977129
// MI455X (gfx1250) — matrix operand order repaired and checked by reading, not compiled, not run on hardware
//
#include <hip/hip_runtime.h>

// ---------------------------------------------------------------------------
// Types for CDNA5 WMMA (wave32, 16x16x32 bf16 -> f32)
// ---------------------------------------------------------------------------
typedef __bf16 v16bf __attribute__((ext_vector_type(16)));
typedef float  v8f   __attribute__((ext_vector_type(8)));
typedef unsigned short u16;
typedef unsigned int   u32;

union BF16Frag {
    uint4 u[2];   // 32 bytes = 16 bf16
    v16bf v;
};

// round-half-up f32 -> bf16 (bias << bf16 ulp; 2 VALU ops)
__device__ __forceinline__ u16 f2bf(float f) {
    return (u16)((__builtin_bit_cast(u32, f) + 0x8000u) >> 16);
}
// pack two f32 -> two bf16 in one dword: lo = a, hi = b (2 adds + 1 v_perm_b32)
__device__ __forceinline__ u32 pack_bf16(float a, float b) {
    u32 ua = __builtin_bit_cast(u32, a) + 0x8000u;
    u32 ub = __builtin_bit_cast(u32, b) + 0x8000u;
    // v_perm_b32: sel bytes 0-3 pick from S1(=ua), 4-7 from S0(=ub)
    return __builtin_amdgcn_perm(ub, ua, 0x07060302u);
}

#define BATCH  65536
#define HID    1024
#define MB     64          // batch rows per block
#define HPAD   1032        // padded K stride for h in LDS (conflict-free ds_b128)
#define FSPAD  72          // padded stride for per-wave feat scratch (64 cols + pad)
#define NWAVES 8

// Dynamic LDS layout (bytes)
#define LDS_H      0
#define LDS_FS     (MB * HPAD * 2)                     // 132096
#define LDS_OA     (LDS_FS + NWAVES * MB * FSPAD * 2)  // 205824
#define LDS_S      (LDS_OA + MB * 32 * 4)              // 214016
#define LDS_SK     (LDS_S + MB * 2 * 4)                // 214528
#define LDS_TOTAL  (LDS_SK + MB * 4)                   // 214784

// ---------------------------------------------------------------------------
// Kernel 0: W2 [k][n] f32 -> W2t [n][k] bf16 ; W_heads [5][1024][4] f32 ->
//           WhT [32][1024] bf16 (row n = skill*4+action, rows 20..31 zero)
// ---------------------------------------------------------------------------
__global__ void convert_weights(const float* __restrict__ W2,
                                const float* __restrict__ W_heads,
                                u16* __restrict__ W2t,
                                u16* __restrict__ WhT) {
    int idx = blockIdx.x * 256 + threadIdx.x;
    if (idx < HID * HID) {
        int n = idx >> 10, k = idx & 1023;
        W2t[idx] = f2bf(W2[k * HID + n]);
    } else {
        int j = idx - HID * HID;          // 0 .. 32*1024-1
        int n = j >> 10, k = j & 1023;
        float v = (n < 20) ? W_heads[(n >> 2) * HID * 4 + k * 4 + (n & 3)] : 0.0f;
        WhT[j] = f2bf(v);
    }
}

// ---------------------------------------------------------------------------
// Kernel 1: fully fused policy net
// ---------------------------------------------------------------------------
__global__ void __launch_bounds__(256)
fused_policy(const float* __restrict__ s,        // [B][2]
             const int*   __restrict__ skill,    // [B]
             const float* __restrict__ W1,       // [2][1024]
             const float* __restrict__ b1,       // [1024]
             const float* __restrict__ b2,       // [1024]
             const u16*   __restrict__ W2t,      // [1024][1024] bf16, [n][k]
             const u16*   __restrict__ WhT,      // [32][1024]   bf16, [n][j]
             const float* __restrict__ b_heads,  // [5][4]
             float*       __restrict__ out) {    // [B][4]
    extern __shared__ __align__(16) char smem[];
    u16*   h_sh  = (u16*)(smem + LDS_H);     // [MB][HPAD] bf16
    u16*   fs_sh = (u16*)(smem + LDS_FS);    // [NWAVES][MB][FSPAD] bf16
    float* oa_sh = (float*)(smem + LDS_OA);  // [MB][32]
    float* s_sh  = (float*)(smem + LDS_S);   // [MB][2]
    int*   sk_sh = (int*)(smem + LDS_SK);    // [MB]

    const int tid  = threadIdx.x;
    const int lane = tid & 31;
    const int wave = tid >> 5;
    const int hi   = lane >> 4;        // 0 or 1 (lane half)
    const int ln   = lane & 15;
    const int row0 = blockIdx.x * MB;

    // ---- stage s / skill, zero head accum plane ----
    if (tid < MB * 2) s_sh[tid] = s[row0 * 2 + tid];
    if (tid < MB)     sk_sh[tid] = skill[row0 + tid];
    for (int i = tid; i < MB * 32; i += 256) oa_sh[i] = 0.0f;
    __syncthreads();

    // ---- layer 1: h = relu(s@W1 + b1) -> bf16 LDS. thread t covers k = 4t..4t+3
    {
        const int k0 = tid * 4;
        float w0[4], w1v[4], bb[4];
        #pragma unroll
        for (int q = 0; q < 4; ++q) {
            w0[q]  = W1[k0 + q];
            w1v[q] = W1[HID + k0 + q];
            bb[q]  = b1[k0 + q];
        }
        for (int m = 0; m < MB; ++m) {
            float s0 = s_sh[2 * m], s1v = s_sh[2 * m + 1];
            float v[4];
            #pragma unroll
            for (int q = 0; q < 4; ++q)
                v[q] = fmaxf(fmaf(s0, w0[q], fmaf(s1v, w1v[q], bb[q])), 0.0f);
            uint2 p;
            p.x = pack_bf16(v[0], v[1]);
            p.y = pack_bf16(v[2], v[3]);
            *(uint2*)&h_sh[m * HPAD + k0] = p;
        }
    }
    __syncthreads();

    // ---- layer 2 + head: wave owns 128 columns, 2 chunks of 64 (4 mt x 4 nt) ----
    const int colw = wave * 128;
    u16* fsw = fs_sh + wave * MB * FSPAD;

    v8f hacc[4][2] = {};   // head accumulators: out_all[64][32] partials

    for (int nc = 0; nc < 2; ++nc) {
        const int colbase = colw + nc * 64;
        v8f acc[4][4] = {};

        // B base: W2t[(colbase + nt*16 + ln)][kt*32 + hi*16]
        const u16* bp = W2t + (size_t)(colbase + ln) * HID + (hi << 4);

        // ---- software-pipelined K loop: double-buffered B fragments ----
        BF16Frag bf[2][4];
        #pragma unroll
        for (int nt = 0; nt < 4; ++nt) {
            const u16* p = bp + nt * 16 * HID;
            bf[0][nt].u[0] = *(const uint4*)p;
            bf[0][nt].u[1] = *(const uint4*)(p + 8);
        }

        for (int kt = 0; kt < 32; ++kt) {
            const int cur = kt & 1, nxt = cur ^ 1;
            if (kt + 1 < 32) {           // issue next k-tile's B loads now
                #pragma unroll
                for (int nt = 0; nt < 4; ++nt) {
                    const u16* p = bp + nt * 16 * HID + (kt + 1) * 32;
                    bf[nxt][nt].u[0] = *(const uint4*)p;
                    bf[nxt][nt].u[1] = *(const uint4*)(p + 8);
                }
            }
            if (kt + 6 < 32) {           // stream-ahead into the WGP cache
                #pragma unroll
                for (int nt = 0; nt < 4; ++nt)
                    __builtin_prefetch(bp + nt * 16 * HID + (kt + 6) * 32, 0, 3);
            }
            // preload all 4 A fragments so waits can be staged
            BF16Frag af[4];
            #pragma unroll
            for (int mt = 0; mt < 4; ++mt) {
                const u16* ap = h_sh + (mt * 16 + ln) * HPAD + kt * 32 + (hi << 4);
                af[mt].u[0] = *(const uint4*)ap;
                af[mt].u[1] = *(const uint4*)(ap + 8);
            }
            #pragma unroll
            for (int mt = 0; mt < 4; ++mt)
                #pragma unroll
                for (int nt = 0; nt < 4; ++nt)
                    acc[mt][nt] = __builtin_amdgcn_wmma_f32_16x16x32_bf16(
                        false, af[mt].v, false, bf[cur][nt].v, (short)0,
                        acc[mt][nt], false, false);
        }

        // ---- epilogue: bias + relu -> bf16 feat chunk into wave scratch ----
        #pragma unroll
        for (int nt = 0; nt < 4; ++nt) {
            float bv = b2[colbase + nt * 16 + ln];
            #pragma unroll
            for (int mt = 0; mt < 4; ++mt) {
                #pragma unroll
                for (int r = 0; r < 8; ++r) {
                    float v = fmaxf(acc[mt][nt][r] + bv, 0.0f);
                    int m = mt * 16 + r + (hi << 3);
                    fsw[m * FSPAD + nt * 16 + ln] = f2bf(v);
                }
            }
        }

        // ---- head GEMM for this chunk (K=64 -> 2 k-tiles) ----
        #pragma unroll
        for (int hkt = 0; hkt < 2; ++hkt) {
            BF16Frag hb0, hb1;
            const u16* hb = WhT + (size_t)ln * HID + colbase + hkt * 32 + (hi << 4);
            hb0.u[0] = *(const uint4*)hb;
            hb0.u[1] = *(const uint4*)(hb + 8);
            hb1.u[0] = *(const uint4*)(hb + 16 * HID);
            hb1.u[1] = *(const uint4*)(hb + 16 * HID + 8);
            #pragma unroll
            for (int mt = 0; mt < 4; ++mt) {
                BF16Frag fa;
                const u16* fap = fsw + (mt * 16 + ln) * FSPAD + hkt * 32 + (hi << 4);
                fa.u[0] = *(const uint4*)fap;
                fa.u[1] = *(const uint4*)(fap + 8);
                hacc[mt][0] = __builtin_amdgcn_wmma_f32_16x16x32_bf16(
                    false, fa.v, false, hb0.v, (short)0, hacc[mt][0], false, false);
                hacc[mt][1] = __builtin_amdgcn_wmma_f32_16x16x32_bf16(
                    false, fa.v, false, hb1.v, (short)0, hacc[mt][1], false, false);
            }
        }
    }

    // ---- cross-wave reduction of head partials (conflict-free ds_add_f32) ----
    #pragma unroll
    for (int mt = 0; mt < 4; ++mt)
        #pragma unroll
        for (int nt = 0; nt < 2; ++nt)
            #pragma unroll
            for (int r = 0; r < 8; ++r) {
                int m = mt * 16 + r + (hi << 3);
                int n = nt * 16 + ln;
                atomicAdd(&oa_sh[m * 32 + n], hacc[mt][nt][r]);
            }
    __syncthreads();

    // ---- per-row skill select + head bias + store: thread = (row, action) ----
    {
        int m = tid >> 2, a = tid & 3;
        int sk = sk_sh[m];
        float v = oa_sh[m * 32 + sk * 4 + a] + b_heads[sk * 4 + a];
        out[(size_t)(row0 + m) * 4 + a] = v;
    }
}

// ---------------------------------------------------------------------------
// Host launch
// ---------------------------------------------------------------------------
extern "C" void kernel_launch(void* const* d_in, const int* in_sizes, int n_in,
                              void* d_out, int out_size, void* d_ws, size_t ws_size,
                              hipStream_t stream) {
    const float* s       = (const float*)d_in[0];
    const int*   skill   = (const int*)d_in[1];
    const float* W1      = (const float*)d_in[2];
    const float* b1      = (const float*)d_in[3];
    const float* W2      = (const float*)d_in[4];
    const float* b2      = (const float*)d_in[5];
    const float* W_heads = (const float*)d_in[6];
    const float* b_heads = (const float*)d_in[7];
    float* out = (float*)d_out;

    u16* W2t = (u16*)d_ws;                                  // 2 MB
    u16* WhT = (u16*)((char*)d_ws + (size_t)HID * HID * 2); // 64 KB

    // weight transpose + bf16 conversion (one-time cost per launch)
    convert_weights<<<(HID * HID + 32 * HID) / 256, 256, 0, stream>>>(
        W2, W_heads, W2t, WhT);

    // raise dynamic-LDS cap (CDNA5 WGP has 320 KB); deterministic, no guards
    (void)hipFuncSetAttribute(reinterpret_cast<const void*>(fused_policy),
                              hipFuncAttributeMaxDynamicSharedMemorySize,
                              LDS_TOTAL);

    fused_policy<<<BATCH / MB, 256, LDS_TOTAL, stream>>>(
        s, skill, W1, b1, b2, W2t, WhT, b_heads, out);
}
